// LSTMModel_32899449488173
// MI455X (gfx1250) — compile-verified, matrix operand order checked by reading, not run on hardware
//
#include <hip/hip_runtime.h>

typedef __attribute__((ext_vector_type(16))) _Float16 v16h;
typedef __attribute__((ext_vector_type(8)))  _Float16 v8h;
typedef __attribute__((ext_vector_type(8)))  float    v8f;
typedef __attribute__((ext_vector_type(4)))  float    v4f;

#define TLEN  512     // timesteps
#define HN    64      // hidden size
#define BTILE 16      // batch tile per workgroup
#define XROW  516     // padded x row stride (f32) -> conflict-free broadcasts
#define HROW  72      // padded h row stride (f16) -> conflict-light A loads

union Frag16 { v16h v; v8h h[2]; };

__device__ __forceinline__ float fast_sigmoid(float x) {
    return __builtin_amdgcn_rcpf(1.0f + __expf(-x));
}
__device__ __forceinline__ float fast_tanh(float x) {
    // tanh(x) = 2*sigmoid(2x) - 1
    return __builtin_fmaf(2.0f, fast_sigmoid(2.0f * x), -1.0f);
}

__global__ __launch_bounds__(128)
void lstm_wmma_fused(const float* __restrict__ x,      // [B, T] (I==1)
                     const float* __restrict__ w_ih,   // [256, 1]
                     const float* __restrict__ w_hh,   // [256, 64]
                     const float* __restrict__ b_ih,   // [256]
                     const float* __restrict__ b_hh,   // [256]
                     const float* __restrict__ w_lin,  // [1, 64]
                     const float* __restrict__ b_lin,  // [1]
                     float* __restrict__ out)          // [B, 1]
{
    __shared__ float               x_lds[BTILE * XROW];        // ~33 KB, whole x tile
    __shared__ __align__(16) _Float16 h_lds[2][BTILE * HROW];  // ping-pong h state (f16)
    __shared__ float               obuf[BTILE];

    const int tid  = threadIdx.x;
    const int wave = tid >> 5;        // 0..3 : 16 hidden columns each
    const int lane = tid & 31;
    const int l    = lane & 15;
    const int hi   = lane >> 4;
    const int b0   = blockIdx.x * BTILE;

    // ---- stage x tile into LDS (coalesced global reads, one-time) ----
    #pragma unroll 4
    for (int k = 0; k < (BTILE * TLEN) / 128; ++k) {
        int idx = tid + k * 128;
        x_lds[(idx >> 9) * XROW + (idx & 511)] = x[(size_t)b0 * TLEN + idx];
    }
    // ---- zero h state (both buffers) ----
    for (int idx = tid; idx < 2 * BTILE * HROW; idx += 128)
        h_lds[0][idx] = (_Float16)0.0f;
    if (tid < BTILE) obuf[tid] = 0.0f;

    // ---- this lane's hidden column ----
    const int j = wave * 16 + l;      // 0..63

    // ---- B fragments: w_hh^T columns for all 4 gates of column j, kept in VGPRs ----
    Frag16 Bf[4][2];
    #pragma unroll
    for (int gate = 0; gate < 4; ++gate) {
        #pragma unroll
        for (int kb = 0; kb < 2; ++kb) {          // K blocks 0..31, 32..63
            #pragma unroll
            for (int c = 0; c < 2; ++c) {         // fragment halves 0-7 / 8-15
                const float* src = w_hh + (size_t)(gate * 64 + j) * HN
                                        + kb * 32 + c * 16 + hi * 8;
                v4f f0 = *(const v4f*)(src);
                v4f f1 = *(const v4f*)(src + 4);
                v8h hh;
                #pragma unroll
                for (int p = 0; p < 4; ++p) {
                    hh[p]     = (_Float16)f0[p];
                    hh[4 + p] = (_Float16)f1[p];
                }
                Bf[gate][kb].h[c] = hh;
            }
        }
    }

    // ---- per-lane gate constants ----
    const float wi_i = w_ih[j],       wi_f = w_ih[64 + j],
                wi_g = w_ih[128 + j], wi_o = w_ih[192 + j];
    const float bc_i = b_ih[j]       + b_hh[j];
    const float bc_f = b_ih[64 + j]  + b_hh[64 + j];
    const float bc_g = b_ih[128 + j] + b_hh[128 + j];
    const float bc_o = b_ih[192 + j] + b_hh[192 + j];
    const float wl   = w_lin[j];

    float cst[8], hlast[8];
    #pragma unroll
    for (int r = 0; r < 8; ++r) { cst[r] = 0.0f; hlast[r] = 0.0f; }

    __syncthreads();

    #pragma unroll 1
    for (int t = 0; t < TLEN; ++t) {
        const _Float16* hr = &h_lds[t & 1][l * HROW];        // read buffer
        _Float16*       hw = &h_lds[(t + 1) & 1][0];         // write buffer

        // ---- A fragments: h[16 x 64] f16 (shared across the 4 gate tiles) ----
        Frag16 A0, A1;
        A0.h[0] = *(const v8h*)(hr + hi * 8);
        A0.h[1] = *(const v8h*)(hr + 16 + hi * 8);
        A1.h[0] = *(const v8h*)(hr + 32 + hi * 8);
        A1.h[1] = *(const v8h*)(hr + 48 + hi * 8);

        // ---- 4 gate tiles: [16b x 64] x [64 x 16], f32 accumulate ----
        v8f ai = {}, af = {}, ag = {}, ao = {};
        ai = __builtin_amdgcn_wmma_f32_16x16x32_f16(false, A0.v, false, Bf[0][0].v, (short)0, ai, false, false);
        af = __builtin_amdgcn_wmma_f32_16x16x32_f16(false, A0.v, false, Bf[1][0].v, (short)0, af, false, false);
        ag = __builtin_amdgcn_wmma_f32_16x16x32_f16(false, A0.v, false, Bf[2][0].v, (short)0, ag, false, false);
        ao = __builtin_amdgcn_wmma_f32_16x16x32_f16(false, A0.v, false, Bf[3][0].v, (short)0, ao, false, false);
        ai = __builtin_amdgcn_wmma_f32_16x16x32_f16(false, A1.v, false, Bf[0][1].v, (short)0, ai, false, false);
        af = __builtin_amdgcn_wmma_f32_16x16x32_f16(false, A1.v, false, Bf[1][1].v, (short)0, af, false, false);
        ag = __builtin_amdgcn_wmma_f32_16x16x32_f16(false, A1.v, false, Bf[2][1].v, (short)0, ag, false, false);
        ao = __builtin_amdgcn_wmma_f32_16x16x32_f16(false, A1.v, false, Bf[3][1].v, (short)0, ao, false, false);

        // ---- gate math fully in registers: 8 cells per lane ----
        #pragma unroll
        for (int r = 0; r < 8; ++r) {
            const int m = r + 8 * hi;                     // batch row of this cell
            const float xv = x_lds[m * XROW + t];         // broadcast read
            float ii = fast_sigmoid(ai[r] + xv * wi_i + bc_i);
            float ff = fast_sigmoid(af[r] + xv * wi_f + bc_f);
            float gg = fast_tanh   (ag[r] + xv * wi_g + bc_g);
            float oo = fast_sigmoid(ao[r] + xv * wi_o + bc_o);
            cst[r]   = ff * cst[r] + ii * gg;
            hlast[r] = oo * fast_tanh(cst[r]);
            hw[m * HROW + j] = (_Float16)hlast[r];
        }

        __syncthreads();   // single barrier: new-h visible before next A loads
    }

    // ---- out[b] = h_last . w_lin + b_lin (f32 h still in registers) ----
    #pragma unroll
    for (int r = 0; r < 8; ++r)
        atomicAdd(&obuf[r + 8 * hi], hlast[r] * wl);
    __syncthreads();
    if (tid < BTILE) out[b0 + tid] = obuf[tid] + b_lin[0];
}

extern "C" void kernel_launch(void* const* d_in, const int* in_sizes, int n_in,
                              void* d_out, int out_size, void* d_ws, size_t ws_size,
                              hipStream_t stream) {
    const float* x     = (const float*)d_in[0];
    const float* w_ih  = (const float*)d_in[1];
    const float* w_hh  = (const float*)d_in[2];
    const float* b_ih  = (const float*)d_in[3];
    const float* b_hh  = (const float*)d_in[4];
    const float* w_lin = (const float*)d_in[5];
    const float* b_lin = (const float*)d_in[6];
    float* out = (float*)d_out;

    const int B = 2048;
    dim3 grid(B / BTILE);   // 128 workgroups (one per 16-batch M-tile)
    dim3 block(128);        // 4 waves, each owns 16 hidden cols x 4 gates
    lstm_wmma_fused<<<grid, block, 0, stream>>>(x, w_ih, w_hh, b_ih, b_hh,
                                                w_lin, b_lin, out);
}
